// CitationNet_90417651516148
// MI455X (gfx1250) — matrix operand order repaired and checked by reading, not compiled, not run on hardware
//
#include <hip/hip_runtime.h>

// ---------------------------------------------------------------------------
// CitationNet EGAT pipeline for MI455X (gfx1250, wave32, WMMA)
// Node-side projections: fp16 WMMA GEMMs (v_wmma_f32_16x16x32_f16),
// software-pipelined, 16x32 C-tile per wave (2 WMMAs / K-step).
// Edge-side attention/scatter: fp32 VALU + global atomics (L2-resident).
// ---------------------------------------------------------------------------

typedef __attribute__((ext_vector_type(16))) _Float16 v16h;
typedef __attribute__((ext_vector_type(8)))  float    v8f;

union FragH { v16h h; uint4 q[2]; };

static constexpr int FIN  = 128;
static constexpr int FV   = 256;
static constexpr int FE   = 64;
static constexpr int FEIN = 16;
static constexpr int NC   = 40;
static constexpr int H    = 8;
static constexpr int DV   = 16;     // per-head node dim
static constexpr int HNCH = H * NC; // 320

// ----------------------------- utility kernels -----------------------------

__global__ void fill_kernel(float* __restrict__ p, float v, long long n) {
  long long i = (long long)blockIdx.x * blockDim.x + threadIdx.x;
  if (i < n) p[i] = v;
}

__global__ void cast_f32_f16(const float* __restrict__ in, _Float16* __restrict__ out, long long n) {
  long long i = (long long)blockIdx.x * blockDim.x + threadIdx.x;
  if (i < n) out[i] = (_Float16)in[i];
}

// W [K, Ncol] f32  ->  Wt [Ncol, K] f16   (B^T for the TN GEMM)
__global__ void transpose_cast(const float* __restrict__ W, _Float16* __restrict__ Wt,
                               int K, int Ncol) {
  int i = blockIdx.x * blockDim.x + threadIdx.x;
  if (i < K * Ncol) {
    int k = i / Ncol, n = i % Ncol;
    Wt[(long long)n * K + k] = (_Float16)W[i];
  }
}

// --------------------------- WMMA GEMM (TN, fp16) --------------------------
// C[M,Ncol] f32 = A[M,K] f16 (row major) x Bt[Ncol,K] f16 (row major, = B^T)
// One wave per 16x32 C tile (two 16x16 WMMA tiles sharing the A fragment).
// K must be a multiple of 32; Ncol a multiple of 32.
// A/B fragments use contiguous k order (k = grp*16 + s) in both operands (16x32 A, 32x16 B).
__global__ __launch_bounds__(32) void gemm_tn_wmma(
    const _Float16* __restrict__ A, const _Float16* __restrict__ Bt,
    float* __restrict__ C, int M, int Ncol, int K)
{
  const int lane = threadIdx.x;
  const int grp  = lane >> 4;      // 0/1  (half-wave)
  const int l15  = lane & 15;

  int m = blockIdx.x * 16 + l15;
  int mload = m < M ? m : M - 1;   // clamp for ragged M (stores are guarded)
  const _Float16* arow  = A  + (long long)mload * K;
  const _Float16* brow0 = Bt + (long long)(blockIdx.y * 32 + l15) * K;
  const _Float16* brow1 = brow0 + (long long)16 * K;

  v8f acc0 = {}, acc1 = {};
  FragH a, b0, b1;
  // --- prologue: load K-step 0 fragments ---
  // A frag: dword j -> K = grp*16 + 2j, 2j+1 (16 contiguous halfs)
  a.q[0]  = *(const uint4*)(arow + grp * 16);
  a.q[1]  = *(const uint4*)(arow + grp * 16 + 8);
  // B frag: dword j -> K = grp*16 + 2j, 2j+1 (16 contiguous halfs)
  b0.q[0] = *(const uint4*)(brow0 + grp * 16);
  b0.q[1] = *(const uint4*)(brow0 + grp * 16 + 8);
  b1.q[0] = *(const uint4*)(brow1 + grp * 16);
  b1.q[1] = *(const uint4*)(brow1 + grp * 16 + 8);

  // --- software-pipelined main loop: issue next loads, then WMMA current ---
  for (int k0 = 32; k0 < K; k0 += 32) {
    FragH an, b0n, b1n;
    an.q[0]  = *(const uint4*)(arow + k0 + grp * 16);
    an.q[1]  = *(const uint4*)(arow + k0 + grp * 16 + 8);
    b0n.q[0] = *(const uint4*)(brow0 + k0 + grp * 16);
    b0n.q[1] = *(const uint4*)(brow0 + k0 + grp * 16 + 8);
    b1n.q[0] = *(const uint4*)(brow1 + k0 + grp * 16);
    b1n.q[1] = *(const uint4*)(brow1 + k0 + grp * 16 + 8);
    acc0 = __builtin_amdgcn_wmma_f32_16x16x32_f16(false, a.h, false, b0.h,
                                                  (short)0, acc0, false, false);
    acc1 = __builtin_amdgcn_wmma_f32_16x16x32_f16(false, a.h, false, b1.h,
                                                  (short)0, acc1, false, false);
    a = an; b0 = b0n; b1 = b1n;
  }
  acc0 = __builtin_amdgcn_wmma_f32_16x16x32_f16(false, a.h, false, b0.h,
                                                (short)0, acc0, false, false);
  acc1 = __builtin_amdgcn_wmma_f32_16x16x32_f16(false, a.h, false, b1.h,
                                                (short)0, acc1, false, false);

  // C frag: VGPR r -> row = tile_m*16 + grp*8 + r, col = tile_n + l15
  int rbase = blockIdx.x * 16 + grp * 8;
  float* crow = C + (long long)rbase * Ncol + blockIdx.y * 32 + l15;
#pragma unroll
  for (int r = 0; r < 8; ++r) {
    if (rbase + r < M) {
      crow[(long long)r * Ncol]      = acc0[r];
      crow[(long long)r * Ncol + 16] = acc1[r];
    }
  }
}

// ------------------------------ edge MLP -----------------------------------
// eout[e,:64] = relu( sp[src[e]] + dp[dst[e]] + ein[e] @ We + b )
// Safe for eout == ein (row staged in LDS before overwrite).
template <int KIN>
__global__ __launch_bounds__(256) void edge_mlp_kernel(
    const float* __restrict__ ein, const int* __restrict__ src, const int* __restrict__ dst,
    const float* __restrict__ sp, const float* __restrict__ dp,
    const float* __restrict__ We, const float* __restrict__ b,
    float* __restrict__ eout, int E_)
{
  __shared__ float sWe[KIN * 64];
  __shared__ float sb[64];
  __shared__ float srow[4][64];
  for (int i = threadIdx.x; i < KIN * 64; i += 256) sWe[i] = We[i];
  if (threadIdx.x < 64) sb[threadIdx.x] = b[threadIdx.x];

  const int j    = threadIdx.x & 63;
  const int esub = threadIdx.x >> 6;
  const int e0   = blockIdx.x * 64;

  for (int it = 0; it < 16; ++it) {
    int e = e0 + it * 4 + esub;
    bool valid = e < E_;
    __syncthreads();
    if (valid && j < KIN) srow[esub][j] = ein[(long long)e * KIN + j];
    __syncthreads();
    if (valid) {
      float acc = sb[j] + sp[(long long)src[e] * 64 + j] + dp[(long long)dst[e] * 64 + j];
#pragma unroll 8
      for (int k = 0; k < KIN; ++k) acc += srow[esub][k] * sWe[k * 64 + j];
      eout[(long long)e * 64 + j] = acc > 0.f ? acc : 0.f;
    }
  }
}

// --------------------------- attention helpers -----------------------------

__device__ inline void atomicMaxFloat(float* addr, float val) {
  if (val >= 0.f) atomicMax((int*)addr, __float_as_int(val));
  else            atomicMin((unsigned int*)addr, __float_as_uint(val));
}

// per-node attention dots: dsn/ddn[n,h] = <feat[n,h,:D], a_s/a_d[h,:D]>
__global__ void node_dots(const float* __restrict__ feat,
                          const float* __restrict__ a_s, const float* __restrict__ a_d,
                          float* __restrict__ dsn, float* __restrict__ ddn, int Nn, int D) {
  int i = blockIdx.x * blockDim.x + threadIdx.x;
  if (i >= Nn * H) return;
  int n = i >> 3, h = i & 7;
  const float* v  = feat + (long long)n * H * D + h * D;
  const float* as = a_s + h * D;
  const float* ad = a_d + h * D;
  float s = 0.f, d = 0.f;
  for (int k = 0; k < D; ++k) { s += v[k] * as[k]; d += v[k] * ad[k]; }
  dsn[i] = s; ddn[i] = d;
}

// wae[h,k] = sum_d We[k, h*16+d] * a_e[h,d]   (fold a_e into We)
__global__ void make_wae(const float* __restrict__ We, const float* __restrict__ a_e,
                         float* __restrict__ wae) {
  int i = blockIdx.x * blockDim.x + threadIdx.x;
  if (i >= H * 64) return;
  int h = i >> 6, k = i & 63;
  float s = 0.f;
  for (int d = 0; d < 16; ++d) s += We[k * 128 + h * 16 + d] * a_e[h * 16 + d];
  wae[i] = s;
}

// EGAT logits + segment max  (one thread per (edge, head))
__global__ void egat_logits(const float* __restrict__ e, const int* __restrict__ src,
                            const int* __restrict__ dst, const float* __restrict__ dsn,
                            const float* __restrict__ ddn, const float* __restrict__ wae,
                            float* __restrict__ exbuf, float* __restrict__ segmax, int E_) {
  int i = blockIdx.x * blockDim.x + threadIdx.x;
  if (i >= E_ * H) return;
  int ed = i >> 3, h = i & 7;
  const float* er = e + (long long)ed * 64;
  const float* w  = wae + h * 64;
  float de = 0.f;
#pragma unroll 8
  for (int k = 0; k < 64; ++k) de += er[k] * w[k];
  int dn = dst[ed];
  float lg = dsn[(long long)src[ed] * H + h] + ddn[(long long)dn * H + h] + de;
  lg = lg > 0.f ? lg : 0.2f * lg;         // leaky_relu(0.2)
  exbuf[i] = lg;
  atomicMaxFloat(&segmax[(long long)dn * H + h], lg);
}

// GAT logits (no edge term)
__global__ void gat_logits(const int* __restrict__ src, const int* __restrict__ dst,
                           const float* __restrict__ dsn, const float* __restrict__ ddn,
                           float* __restrict__ exbuf, float* __restrict__ segmax, int E_) {
  int i = blockIdx.x * blockDim.x + threadIdx.x;
  if (i >= E_ * H) return;
  int ed = i >> 3, h = i & 7;
  int dn = dst[ed];
  float lg = dsn[(long long)src[ed] * H + h] + ddn[(long long)dn * H + h];
  lg = lg > 0.f ? lg : 0.2f * lg;
  exbuf[i] = lg;
  atomicMaxFloat(&segmax[(long long)dn * H + h], lg);
}

// exp(logit - segmax) and segment sum
__global__ void seg_expsum(const int* __restrict__ dst, float* __restrict__ exbuf,
                           const float* __restrict__ segmax, float* __restrict__ segsum, int E_) {
  int i = blockIdx.x * blockDim.x + threadIdx.x;
  if (i >= E_ * H) return;
  int ed = i >> 3, h = i & 7;
  long long idx = (long long)dst[ed] * H + h;
  float ex = __expf(exbuf[i] - segmax[idx]);
  exbuf[i] = ex;
  atomicAdd(&segsum[idx], ex);
}

// alpha-weighted scatters: nv += alpha*xv[src],  S += alpha*e_row
__global__ void egat_scatter(const float* __restrict__ e, const float* __restrict__ xv,
                             const int* __restrict__ src, const int* __restrict__ dst,
                             const float* __restrict__ exbuf, const float* __restrict__ segsum,
                             float* __restrict__ nv, float* __restrict__ S, int E_) {
  int i = blockIdx.x * blockDim.x + threadIdx.x;
  if (i >= E_ * H) return;
  int ed = i >> 3, h = i & 7;
  long long sn = src[ed], dn = dst[ed];
  float alpha = exbuf[i] / (segsum[dn * H + h] + 1e-16f);
  const float* xr = xv + sn * 128 + h * DV;
  float* nvr = nv + dn * 128 + h * DV;
#pragma unroll
  for (int d = 0; d < DV; ++d) atomicAdd(&nvr[d], alpha * xr[d]);
  const float* er = e + (long long)ed * 64;
  float* Sr = S + dn * 512 + h * 64;
#pragma unroll 8
  for (int k = 0; k < 64; ++k) atomicAdd(&Sr[k], alpha * er[k]);
}

// h[n, 0:128] = elu(nv); h[n, 128+jj] = elu(S[n,h,:] @ We[:, jj]); also f16 copy
__global__ __launch_bounds__(256) void egat_finalize(
    const float* __restrict__ nv, const float* __restrict__ S,
    const float* __restrict__ We, float* __restrict__ hout,
    _Float16* __restrict__ hout_h, int Nn)
{
  __shared__ float sWe[64 * 128];                 // 32 KB
  for (int i = threadIdx.x; i < 64 * 128; i += 256) sWe[i] = We[i];
  __syncthreads();
  int j = threadIdx.x;
  int nend = blockIdx.x * 8 + 8; if (nend > Nn) nend = Nn;
  for (int n = blockIdx.x * 8; n < nend; ++n) {
    float v;
    if (j < 128) {
      v = nv[(long long)n * 128 + j];
    } else {
      int jj = j - 128, h = jj >> 4;
      const float* Sr = S + (long long)n * 512 + h * 64;
      float s = 0.f;
#pragma unroll 8
      for (int k = 0; k < 64; ++k) s += Sr[k] * sWe[k * 128 + jj];
      v = s;
    }
    v = v > 0.f ? v : (__expf(v) - 1.f);          // elu
    hout[(long long)n * 256 + j]   = v;
    hout_h[(long long)n * 256 + j] = (_Float16)v;
  }
}

// final GAT scatter: gacc[dst, h*40+d] += alpha * xg[src, h*40+d]
__global__ void gat_scatter(const float* __restrict__ xg, const int* __restrict__ src,
                            const int* __restrict__ dst, const float* __restrict__ exbuf,
                            const float* __restrict__ segsum, float* __restrict__ gacc, int E_) {
  int i = blockIdx.x * blockDim.x + threadIdx.x;
  if (i >= E_ * H) return;
  int ed = i >> 3, h = i & 7;
  long long sn = src[ed], dn = dst[ed];
  float alpha = exbuf[i] / (segsum[dn * H + h] + 1e-16f);
  const float* xr = xg + sn * HNCH + h * NC;
  float* gr = gacc + dn * HNCH + h * NC;
#pragma unroll 8
  for (int d = 0; d < NC; ++d) atomicAdd(&gr[d], alpha * xr[d]);
}

// out[n,c] = mean_h gacc[n, h*40+c] + b[c]
__global__ void gat_finalize(const float* __restrict__ gacc, const float* __restrict__ gb,
                             float* __restrict__ out, int Nn) {
  int i = blockIdx.x * blockDim.x + threadIdx.x;
  if (i >= Nn * NC) return;
  int n = i / NC, c = i % NC;
  float s = 0.f;
#pragma unroll
  for (int h = 0; h < H; ++h) s += gacc[(long long)n * HNCH + h * NC + c];
  out[i] = s * 0.125f + gb[c];
}

// ------------------------------- launcher ----------------------------------

extern "C" void kernel_launch(void* const* d_in, const int* in_sizes, int n_in,
                              void* d_out, int out_size, void* d_ws, size_t ws_size,
                              hipStream_t stream) {
  const float* x        = (const float*)d_in[0];
  const int*   src      = (const int*)d_in[1];
  const float* eattr    = (const float*)d_in[2];
  const float* e1_Ws    = (const float*)d_in[3];
  const float* e1_Wd    = (const float*)d_in[4];
  const float* e1_We    = (const float*)d_in[5];
  const float* e1_b     = (const float*)d_in[6];
  const float* c1_Wv    = (const float*)d_in[7];
  const float* c1_We    = (const float*)d_in[8];
  const float* c1_as    = (const float*)d_in[9];
  const float* c1_ad    = (const float*)d_in[10];
  const float* c1_ae    = (const float*)d_in[11];
  const float* e2_Ws    = (const float*)d_in[12];
  const float* e2_Wd    = (const float*)d_in[13];
  const float* e2_We    = (const float*)d_in[14];
  const float* e2_b     = (const float*)d_in[15];
  const float* c2_Wv    = (const float*)d_in[16];
  const float* c2_We    = (const float*)d_in[17];
  const float* c2_as    = (const float*)d_in[18];
  const float* c2_ad    = (const float*)d_in[19];
  const float* c2_ae    = (const float*)d_in[20];
  const float* g_W      = (const float*)d_in[21];
  const float* g_as     = (const float*)d_in[22];
  const float* g_ad     = (const float*)d_in[23];
  const float* g_b      = (const float*)d_in[24];

  const int Nn = in_sizes[0] / FIN;     // 50000
  const int E_ = in_sizes[1] / 2;       // 800000
  const int* dst = src + E_;

  // ---- workspace bump allocator (256B aligned) ----
  char* wp = (char*)d_ws;
  auto alloc = [&](size_t bytes) -> void* {
    void* p = (void*)wp;
    wp += (bytes + 255) & ~(size_t)255;
    return p;
  };
  _Float16* x_h     = (_Float16*)alloc((size_t)Nn * FIN * 2);
  _Float16* e1Ws_t  = (_Float16*)alloc(64 * 128 * 2);
  _Float16* e1Wd_t  = (_Float16*)alloc(64 * 128 * 2);
  _Float16* c1Wv_t  = (_Float16*)alloc(128 * 128 * 2);
  _Float16* e2Ws_t  = (_Float16*)alloc(64 * 256 * 2);
  _Float16* e2Wd_t  = (_Float16*)alloc(64 * 256 * 2);
  _Float16* c2Wv_t  = (_Float16*)alloc(128 * 256 * 2);
  _Float16* gW_t    = (_Float16*)alloc(320 * 256 * 2);
  float* sproj  = (float*)alloc((size_t)Nn * 64 * 4);
  float* dproj  = (float*)alloc((size_t)Nn * 64 * 4);
  float* e_buf  = (float*)alloc((size_t)E_ * 64 * 4);
  float* xv     = (float*)alloc((size_t)Nn * 128 * 4);
  float* dsn    = (float*)alloc((size_t)Nn * H * 4);
  float* ddn    = (float*)alloc((size_t)Nn * H * 4);
  float* wae    = (float*)alloc(H * 64 * 4);
  float* exbuf  = (float*)alloc((size_t)E_ * H * 4);
  float* segmax = (float*)alloc((size_t)Nn * H * 4);
  float* segsum = (float*)alloc((size_t)Nn * H * 4);
  float* nv     = (float*)alloc((size_t)Nn * 128 * 4);
  float* S      = (float*)alloc((size_t)Nn * 512 * 4);
  float* h1     = (float*)alloc((size_t)Nn * 256 * 4);
  _Float16* h1_h = (_Float16*)alloc((size_t)Nn * 256 * 2);
  float* h2     = (float*)alloc((size_t)Nn * 256 * 4);
  _Float16* h2_h = (_Float16*)alloc((size_t)Nn * 256 * 2);
  float* xg     = (float*)alloc((size_t)Nn * HNCH * 4);
  float* gacc   = (float*)alloc((size_t)Nn * HNCH * 4);
  (void)ws_size; (void)n_in; (void)out_size;

  const float NEG_INF = -3.4e38f;
  auto blocks = [](long long n, int bs) { return (unsigned)((n + bs - 1) / bs); };
  auto gemm = [&](const _Float16* A, const _Float16* Bt, float* C, int M, int Ncol, int K) {
    dim3 g((M + 15) / 16, Ncol / 32);
    gemm_tn_wmma<<<g, dim3(32), 0, stream>>>(A, Bt, C, M, Ncol, K);
  };
  auto fill = [&](float* p, float v, long long n) {
    fill_kernel<<<blocks(n, 256), 256, 0, stream>>>(p, v, n);
  };

  // ---------------- prep: fp16 casts + weight transposes ----------------
  cast_f32_f16<<<blocks((long long)Nn * FIN, 256), 256, 0, stream>>>(x, x_h, (long long)Nn * FIN);
  transpose_cast<<<blocks(128 * 64, 256), 256, 0, stream>>>(e1_Ws, e1Ws_t, 128, 64);
  transpose_cast<<<blocks(128 * 64, 256), 256, 0, stream>>>(e1_Wd, e1Wd_t, 128, 64);
  transpose_cast<<<blocks(128 * 128, 256), 256, 0, stream>>>(c1_Wv, c1Wv_t, 128, 128);
  transpose_cast<<<blocks(256 * 64, 256), 256, 0, stream>>>(e2_Ws, e2Ws_t, 256, 64);
  transpose_cast<<<blocks(256 * 64, 256), 256, 0, stream>>>(e2_Wd, e2Wd_t, 256, 64);
  transpose_cast<<<blocks(256 * 128, 256), 256, 0, stream>>>(c2_Wv, c2Wv_t, 256, 128);
  transpose_cast<<<blocks(256 * 320, 256), 256, 0, stream>>>(g_W, gW_t, 256, 320);

  const unsigned ebl = blocks((long long)E_ * H, 256);

  // ---------------- layer 1: edge MLP + EGAT ----------------
  gemm(x_h, e1Ws_t, sproj, Nn, 64, FIN);
  gemm(x_h, e1Wd_t, dproj, Nn, 64, FIN);
  edge_mlp_kernel<FEIN><<<blocks(E_, 64), 256, 0, stream>>>(eattr, src, dst, sproj, dproj,
                                                            e1_We, e1_b, e_buf, E_);
  gemm(x_h, c1Wv_t, xv, Nn, 128, FIN);
  node_dots<<<blocks((long long)Nn * H, 256), 256, 0, stream>>>(xv, c1_as, c1_ad, dsn, ddn, Nn, DV);
  make_wae<<<2, 256, 0, stream>>>(c1_We, c1_ae, wae);
  fill(segmax, NEG_INF, (long long)Nn * H);
  fill(segsum, 0.f, (long long)Nn * H);
  egat_logits<<<ebl, 256, 0, stream>>>(e_buf, src, dst, dsn, ddn, wae, exbuf, segmax, E_);
  seg_expsum<<<ebl, 256, 0, stream>>>(dst, exbuf, segmax, segsum, E_);
  fill(nv, 0.f, (long long)Nn * 128);
  fill(S, 0.f, (long long)Nn * 512);
  egat_scatter<<<ebl, 256, 0, stream>>>(e_buf, xv, src, dst, exbuf, segsum, nv, S, E_);
  egat_finalize<<<blocks(Nn, 8), 256, 0, stream>>>(nv, S, c1_We, h1, h1_h, Nn);

  // ---------------- layer 2: edge MLP (in place) + EGAT ----------------
  gemm(h1_h, e2Ws_t, sproj, Nn, 64, FV);
  gemm(h1_h, e2Wd_t, dproj, Nn, 64, FV);
  edge_mlp_kernel<FE><<<blocks(E_, 64), 256, 0, stream>>>(e_buf, src, dst, sproj, dproj,
                                                          e2_We, e2_b, e_buf, E_);
  gemm(h1_h, c2Wv_t, xv, Nn, 128, FV);
  node_dots<<<blocks((long long)Nn * H, 256), 256, 0, stream>>>(xv, c2_as, c2_ad, dsn, ddn, Nn, DV);
  make_wae<<<2, 256, 0, stream>>>(c2_We, c2_ae, wae);
  fill(segmax, NEG_INF, (long long)Nn * H);
  fill(segsum, 0.f, (long long)Nn * H);
  egat_logits<<<ebl, 256, 0, stream>>>(e_buf, src, dst, dsn, ddn, wae, exbuf, segmax, E_);
  seg_expsum<<<ebl, 256, 0, stream>>>(dst, exbuf, segmax, segsum, E_);
  fill(nv, 0.f, (long long)Nn * 128);
  fill(S, 0.f, (long long)Nn * 512);
  egat_scatter<<<ebl, 256, 0, stream>>>(e_buf, xv, src, dst, exbuf, segsum, nv, S, E_);
  egat_finalize<<<blocks(Nn, 8), 256, 0, stream>>>(nv, S, c2_We, h2, h2_h, Nn);

  // ---------------- final GAT (mean over heads) ----------------
  gemm(h2_h, gW_t, xg, Nn, HNCH, FV);
  node_dots<<<blocks((long long)Nn * H, 256), 256, 0, stream>>>(xg, g_as, g_ad, dsn, ddn, Nn, NC);
  fill(segmax, NEG_INF, (long long)Nn * H);
  fill(segsum, 0.f, (long long)Nn * H);
  gat_logits<<<ebl, 256, 0, stream>>>(src, dst, dsn, ddn, exbuf, segmax, E_);
  seg_expsum<<<ebl, 256, 0, stream>>>(dst, exbuf, segmax, segsum, E_);
  fill(gacc, 0.f, (long long)Nn * HNCH);
  gat_scatter<<<ebl, 256, 0, stream>>>(xg, src, dst, exbuf, segsum, gacc, E_);
  gat_finalize<<<blocks((long long)Nn * NC, 256), 256, 0, stream>>>(gacc, g_b, (float*)d_out, Nn);
}
